// Net_90675349553707
// MI455X (gfx1250) — matrix operand order repaired and checked by reading, not compiled, not run on hardware
//
#include <hip/hip_runtime.h>
#include <hip/hip_bf16.h>
#include <math.h>

// ---------------------------------------------------------------------------
// Problem constants (from the reference)
// ---------------------------------------------------------------------------
#define BB      8
#define NN64    64
#define EE      192
#define DIN     1024
#define DPROJ   256
#define FF      16
#define NNQ     4096          // N*N
#define SK_TAU  0.05f

typedef __attribute__((ext_vector_type(16))) __bf16 v16bf;
typedef __attribute__((ext_vector_type(8)))  float  v8f;

// row r in [0,4096) of the concatenated projection input
__device__ __forceinline__ const float* row_ptr(const float* x1, const float* x2,
                                                const float* e1, const float* e2, int r) {
    if (r < 512)  return x1 + (size_t)r * DIN;
    if (r < 1024) return x2 + (size_t)(r - 512) * DIN;
    if (r < 2560) return e1 + (size_t)(r - 1024) * DIN;
    return e2 + (size_t)(r - 2560) * DIN;
}

__device__ __forceinline__ void cvt8(v16bf& d, int base, float4 u, float4 v) {
    d[base + 0] = (__bf16)u.x; d[base + 1] = (__bf16)u.y;
    d[base + 2] = (__bf16)u.z; d[base + 3] = (__bf16)u.w;
    d[base + 4] = (__bf16)v.x; d[base + 5] = (__bf16)v.y;
    d[base + 6] = (__bf16)v.z; d[base + 7] = (__bf16)v.w;
}

// ---------------------------------------------------------------------------
// K1: H[4096x256] = X[4096x1024] @ Wp[1024x256] + bp   (bf16 WMMA, f32 acc)
// block = 256 threads = 8 waves; block tile 128x32; wave owns two stacked
// 16x16 tiles sharing one B fragment (2 WMMA / K-step / wave).
// LDS tiles are stored in *fragment order* [tile][lane][16 contiguous bf16]
// so each lane reads its whole v16bf fragment as one 32-byte LDS access.
// ---------------------------------------------------------------------------
__global__ __launch_bounds__(256)
void proj_gemm_kernel(const float* __restrict__ x1, const float* __restrict__ x2,
                      const float* __restrict__ e1, const float* __restrict__ e2,
                      const float* __restrict__ Wp, const float* __restrict__ bp,
                      float* __restrict__ H)
{
    __shared__ alignas(32) __bf16 Af[8][32][16];   // 8 m-tiles  (8 KB)
    __shared__ alignas(32) __bf16 Bf[2][32][16];   // 2 n-tiles  (2 KB)

    const int t    = threadIdx.x;
    const int lane = t & 31;
    const int wave = t >> 5;        // 0..7
    const int wm   = wave >> 1;     // 0..3 -> m-tiles 2*wm, 2*wm+1
    const int wn   = wave & 1;      // 0..1 -> n-tile
    const int rowBase = blockIdx.x * 128;
    const int colBase = blockIdx.y * 32;

    v8f acc0 = {}, acc1 = {};

    for (int k0 = 0; k0 < DIN; k0 += 32) {
        // stage A tile 128x32 -> fragment order; prefetch next K slab
        for (int idx = t; idx < 128 * 32; idx += 256) {
            int r = idx >> 5, k = idx & 31;
            const float* rp = row_ptr(x1, x2, e1, e2, rowBase + r);
            float v = rp[k0 + k];
            // A(m,k): lane = m + 16*((k>>3)&1), slot = (k&7) + 8*(k>>4)
            Af[r >> 4][(r & 15) + (((k >> 3) & 1) << 4)][(k & 7) + ((k >> 4) << 3)] = (__bf16)v;
            if (k0 + 32 < DIN) __builtin_prefetch(rp + k0 + 32 + k, 0, 1);
        }
        // stage B tile 32x32 from Wp -> fragment order
        for (int idx = t; idx < 32 * 32; idx += 256) {
            int k = idx >> 5, n = idx & 31;
            float v = Wp[(size_t)(k0 + k) * DPROJ + colBase + n];
            // B(k,n): lane = (n&15) + 16*((k>>3)&1), slot = (k&7) + 8*(k>>4)
            Bf[n >> 4][(n & 15) + (((k >> 3) & 1) << 4)][(k & 7) + ((k >> 4) << 3)] = (__bf16)v;
        }
        __syncthreads();

        v16bf bfr = *(const v16bf*)(&Bf[wn][lane][0]);          // 32B contiguous
        v16bf a0  = *(const v16bf*)(&Af[2 * wm + 0][lane][0]);
        v16bf a1  = *(const v16bf*)(&Af[2 * wm + 1][lane][0]);
        acc0 = __builtin_amdgcn_wmma_f32_16x16x32_bf16(false, a0, false, bfr,
                                                       (short)0, acc0, false, false);
        acc1 = __builtin_amdgcn_wmma_f32_16x16x32_bf16(false, a1, false, bfr,
                                                       (short)0, acc1, false, false);
        __syncthreads();
    }

    // C/D layout: VGPR r -> M = r + 8*(lane>>4), N = lane&15
    const int n  = lane & 15;
    const int mh = (lane >> 4) * 8;
    const int col = colBase + wn * 16 + n;
    const float bias = bp[col];
    #pragma unroll
    for (int r = 0; r < 8; ++r) {
        H[(size_t)(rowBase + (2 * wm + 0) * 16 + mh + r) * DPROJ + col] = acc0[r] + bias;
        H[(size_t)(rowBase + (2 * wm + 1) * 16 + mh + r) * DPROJ + col] = acc1[r] + bias;
    }
}

// ---------------------------------------------------------------------------
// K2: per-segment column mean / inv-std (batch-norm stats over all rows)
// ---------------------------------------------------------------------------
__global__ __launch_bounds__(256)
void stats_kernel(const float* __restrict__ H, float* __restrict__ stats)
{
    const int seg = blockIdx.x, c = threadIdx.x;
    const int start[4] = {0, 512, 1024, 2560};
    const int cnt[4]   = {512, 512, 1536, 1536};
    float s = 0.f, sq = 0.f;
    for (int r = 0; r < cnt[seg]; ++r) {
        float v = H[(size_t)(start[seg] + r) * DPROJ + c];
        s += v; sq += v * v;
    }
    float mu  = s / (float)cnt[seg];
    float var = sq / (float)cnt[seg] - mu * mu;
    stats[seg * 512 + c]       = mu;
    stats[seg * 512 + 256 + c] = rsqrtf(var + 1e-5f);
}

// ---------------------------------------------------------------------------
// K3: batchnorm + ReLU + per-row L2 normalize (in place on H)
// ---------------------------------------------------------------------------
__global__ __launch_bounds__(256)
void norm_kernel(float* __restrict__ H, const float* __restrict__ stats,
                 const float* __restrict__ gamma, const float* __restrict__ beta)
{
    __shared__ float red[256];
    const int r = blockIdx.x, c = threadIdx.x;
    const int seg = (r < 512) ? 0 : (r < 1024) ? 1 : (r < 2560) ? 2 : 3;
    float v = H[(size_t)r * DPROJ + c];
    v = (v - stats[seg * 512 + c]) * stats[seg * 512 + 256 + c] * gamma[c] + beta[c];
    v = fmaxf(v, 0.f);
    red[c] = v * v;
    __syncthreads();
    for (int s = 128; s > 0; s >>= 1) {
        if (c < s) red[c] += red[c + s];
        __syncthreads();
    }
    float nrm = fmaxf(sqrtf(red[0]), 1e-12f);
    H[(size_t)r * DPROJ + c] = v / nrm;
}

// ---------------------------------------------------------------------------
// K4: one-wave 16x16 tile of  out = scale * X @ Y^T  (K = 256, bf16 WMMA)
// A fragment = two contiguous 8-float runs, B fragment = the same two 8-float runs
// of Y[n,:] -> load as float4 (global_load_b128) and convert in registers.
// ---------------------------------------------------------------------------
__device__ __forceinline__ v8f wave_xyT_tile(const float* __restrict__ X,
                                             const float* __restrict__ Y)
{
    const int lane = threadIdx.x & 31;
    const int m    = lane & 15;
    const int kb   = (lane >> 4) * 8;
    const int koff = (lane >> 4) * 16;
    v8f acc = {};
    #pragma unroll
    for (int k0 = 0; k0 < DPROJ; k0 += 32) {
        const float4* xa = (const float4*)(X + (size_t)m * DPROJ + k0 + kb);
        const float4* xb = (const float4*)(X + (size_t)m * DPROJ + k0 + 16 + kb);
        const float4* yb = (const float4*)(Y + (size_t)m * DPROJ + k0 + kb);
        const float4* yc = (const float4*)(Y + (size_t)m * DPROJ + k0 + 16 + kb);
        float4 A0 = xa[0], A1 = xa[1];
        float4 A2 = xb[0], A3 = xb[1];
        float4 B0 = yb[0], B1 = yb[1], B2 = yc[0], B3 = yc[1];
        v16bf a, b;
        cvt8(a, 0, A0, A1);
        cvt8(a, 8, A2, A3);
        cvt8(b, 0, B0, B1);
        cvt8(b, 8, B2, B3);
        acc = __builtin_amdgcn_wmma_f32_16x16x32_bf16(false, a, false, b,
                                                      (short)0, acc, false, false);
    }
    return acc;
}

__global__ __launch_bounds__(32)
void xyT_kernel(const float* __restrict__ P, int xbase, int ybase, int rows,
                float scale, float* __restrict__ out)
{
    const int b = blockIdx.x, tm = blockIdx.y, tn = blockIdx.z;
    const float* X = P + (size_t)(xbase + b * rows + tm * 16) * DPROJ;
    const float* Y = P + (size_t)(ybase + b * rows + tn * 16) * DPROJ;
    v8f acc = wave_xyT_tile(X, Y);
    const int lane = threadIdx.x & 31;
    const int n  = lane & 15;
    const int mh = (lane >> 4) * 8;
    float* o = out + (size_t)b * rows * rows + (size_t)(tm * 16) * rows + tn * 16;
    #pragma unroll
    for (int r = 0; r < 8; ++r)
        o[(size_t)(mh + r) * rows + n] = acc[r] * scale;
}

// ---------------------------------------------------------------------------
// K5: x0[b,i,0] = Kp^T.ravel()[i] = Kp[b][i%64][i/64]
// ---------------------------------------------------------------------------
__global__ void x0_kernel(const float* __restrict__ Kp, float* __restrict__ x)
{
    const int b = blockIdx.x;
    for (int i = threadIdx.x; i < NNQ; i += blockDim.x)
        x[(size_t)(b * NNQ + i) * 17] = Kp[b * NNQ + (i & 63) * 64 + (i >> 6)];
}

// ---------------------------------------------------------------------------
// K6: one block per (row i, batch b): sparse K-row scatter in LDS, dedup,
//     deg = #(K[i,:] > 0) clipped >= 1, msg = K[i,:] @ x / deg, MLP, v.
// ---------------------------------------------------------------------------
__global__ __launch_bounds__(256)
void msg_mlp_kernel(const float* __restrict__ Ke, const float* __restrict__ Kp,
                    const int* __restrict__ s1, const int* __restrict__ d1,
                    const int* __restrict__ s2, const int* __restrict__ d2,
                    const float* __restrict__ xin, float* __restrict__ xout,
                    float* __restrict__ v3,
                    const float* __restrict__ W, const float* __restrict__ bW,
                    const float* __restrict__ S, const float* __restrict__ cS, int C)
{
    const int i = blockIdx.x, b = blockIdx.y, t = threadIdx.x;
    __shared__ float rowK[NNQ];       // dense K row (16 KB of the 320 KB LDS)
    __shared__ int   listE[EE], listF[EE];
    __shared__ int   nE, nF, degc;
    __shared__ float msg[17];
    __shared__ float hsh[FF];

    for (int j = t; j < NNQ; j += 256) rowK[j] = 0.f;
    if (t == 0) { nE = 0; nF = 0; degc = 0; }
    if (t < 17) msg[t] = 0.f;
    __syncthreads();

    const int a  = i >> 6;    // needs src2[f] == a
    const int bc = i & 63;    // needs src1[e] == bc
    const int* s1b = s1 + b * EE;  const int* d1b = d1 + b * EE;
    const int* s2b = s2 + b * EE;  const int* d2b = d2 + b * EE;

    if (t < EE) {
        if (s1b[t] == bc) { int p = atomicAdd(&nE, 1); listE[p] = t; }
        if (s2b[t] == a)  { int p = atomicAdd(&nF, 1); listF[p] = t; }
    }
    __syncthreads();

    const int npair = nE * nF;
    for (int p = t; p < npair; p += 256) {
        int e = listE[p % nE], f = listF[p / nE];
        int col = d2b[f] * 64 + d1b[e];
        atomicAdd(&rowK[col], Ke[(size_t)(b * EE + e) * EE + f]);   // ds_add_f32
    }
    __syncthreads();

    // dedup pass: atomicExch drains each accumulated cell exactly once
    for (int p = t; p < npair; p += 256) {
        int e = listE[p % nE], f = listF[p / nE];
        int col = d2b[f] * 64 + d1b[e];
        if (col == i) continue;                       // diagonal is overwritten
        float val = atomicExch(&rowK[col], 0.0f);
        if (val != 0.0f) {
            if (val > 0.0f) atomicAdd(&degc, 1);
            const float* xc = xin + (size_t)(b * NNQ + col) * 17;
            for (int c = 0; c < C; ++c) atomicAdd(&msg[c], val * xc[c]);
        }
    }
    if (t == 0) {
        float dval = Kp[b * NNQ + (i & 63) * 64 + (i >> 6)];  // diag := Kp^T
        if (dval > 0.0f) atomicAdd(&degc, 1);
        const float* xc = xin + (size_t)(b * NNQ + i) * 17;
        for (int c = 0; c < C; ++c) atomicAdd(&msg[c], dval * xc[c]);
    }
    __syncthreads();

    const float deg = fmaxf((float)degc, 1.0f);
    if (t < FF) {
        float accv = bW[t];
        for (int c = 0; c < C; ++c) accv += (msg[c] / deg) * W[c * FF + t];
        float h = fmaxf(accv, 0.f);
        hsh[t] = h;
        xout[(size_t)(b * NNQ + i) * 17 + t] = h;
    }
    __syncthreads();
    if (t == 0) {
        float v = cS[0];
        for (int j = 0; j < FF; ++j) v += hsh[j] * S[j];
        v3[b * NNQ + i] = v;
    }
}

// ---------------------------------------------------------------------------
// K7: log-domain Sinkhorn on v^T / tau (64x64, 10 iters); writes x[...,16]
// ---------------------------------------------------------------------------
__global__ __launch_bounds__(64)
void sinkhorn64_kernel(const float* __restrict__ v3, float* __restrict__ xout)
{
    __shared__ float M[64][65];
    const int b = blockIdx.x, t = threadIdx.x;
    for (int c = 0; c < 64; ++c)
        M[t][c] = v3[b * NNQ + c * 64 + t] * (1.0f / SK_TAU);   // transpose
    __syncthreads();
    for (int it = 0; it < 10; ++it) {
        { // axis=2 (over columns), per row t
            float mx = -1e30f;
            for (int c = 0; c < 64; ++c) mx = fmaxf(mx, M[t][c]);
            float s = 0.f;
            for (int c = 0; c < 64; ++c) s += __expf(M[t][c] - mx);
            float l = mx + __logf(s);
            for (int c = 0; c < 64; ++c) M[t][c] -= l;
        }
        __syncthreads();
        { // axis=1 (over rows), per column t
            float mx = -1e30f;
            for (int r = 0; r < 64; ++r) mx = fmaxf(mx, M[r][t]);
            float s = 0.f;
            for (int r = 0; r < 64; ++r) s += __expf(M[r][t] - mx);
            float l = mx + __logf(s);
            for (int r = 0; r < 64; ++r) M[r][t] -= l;
        }
        __syncthreads();
    }
    // x[b, i, 16] = ss^T.ravel()[i] = ss[i%64][i/64]  ->  i = c*64 + t holds ss[t][c]
    for (int c = 0; c < 64; ++c)
        xout[(size_t)(b * NNQ + c * 64 + t) * 17 + 16] = __expf(M[t][c]);
}

// ---------------------------------------------------------------------------
// K8: final readout v = x@Wc+bc, transpose, pad with bin_value -> ls[65x65]
// ---------------------------------------------------------------------------
__global__ __launch_bounds__(256)
void final_v_kernel(const float* __restrict__ x, const float* __restrict__ Wc,
                    const float* __restrict__ bc, const float* __restrict__ binv,
                    float* __restrict__ ls)
{
    const int b = blockIdx.x;
    const float bv = binv[0];
    for (int idx = threadIdx.x; idx < 65 * 65; idx += 256) {
        int r = idx / 65, c = idx % 65;
        float v;
        if (r < 64 && c < 64) {
            const float* xc = x + (size_t)(b * NNQ + c * 64 + r) * 17;  // s = v^T
            v = bc[0];
            for (int ch = 0; ch < 17; ++ch) v += xc[ch] * Wc[ch];
        } else v = bv;
        ls[b * 4225 + idx] = v;
    }
}

// ---------------------------------------------------------------------------
// K9: 65x65 Sinkhorn (10 iters), keep [:64,:64] as exp()
// ---------------------------------------------------------------------------
__global__ __launch_bounds__(96)
void sinkhorn65_kernel(const float* __restrict__ ls, float* __restrict__ ssf)
{
    __shared__ float M[65][66];
    const int b = blockIdx.x, t = threadIdx.x;
    if (t < 65)
        for (int c = 0; c < 65; ++c) M[t][c] = ls[b * 4225 + t * 65 + c];
    __syncthreads();
    for (int it = 0; it < 10; ++it) {
        if (t < 65) {
            float mx = -1e30f;
            for (int c = 0; c < 65; ++c) mx = fmaxf(mx, M[t][c]);
            float s = 0.f;
            for (int c = 0; c < 65; ++c) s += __expf(M[t][c] - mx);
            float l = mx + __logf(s);
            for (int c = 0; c < 65; ++c) M[t][c] -= l;
        }
        __syncthreads();
        if (t < 65) {
            float mx = -1e30f;
            for (int r = 0; r < 65; ++r) mx = fmaxf(mx, M[r][t]);
            float s = 0.f;
            for (int r = 0; r < 65; ++r) s += __expf(M[r][t] - mx);
            float l = mx + __logf(s);
            for (int r = 0; r < 65; ++r) M[r][t] -= l;
        }
        __syncthreads();
    }
    if (t < 64)
        for (int c = 0; c < 64; ++c)
            ssf[b * NNQ + t * 64 + c] = __expf(M[t][c]);
}

// ---------------------------------------------------------------------------
// K10: global max over all batches, then out = ss / max  (single block)
// ---------------------------------------------------------------------------
__global__ __launch_bounds__(256)
void finalize_kernel(const float* __restrict__ ssf, float* __restrict__ out)
{
    __shared__ float red[256];
    const int t = threadIdx.x;
    float mx = -1e30f;
    for (int i = t; i < BB * NNQ; i += 256) mx = fmaxf(mx, ssf[i]);
    red[t] = mx;
    __syncthreads();
    for (int s = 128; s > 0; s >>= 1) {
        if (t < s) red[t] = fmaxf(red[t], red[t + s]);
        __syncthreads();
    }
    const float m = red[0];
    for (int i = t; i < BB * NNQ; i += 256) out[i] = ssf[i] / m;
}

// ---------------------------------------------------------------------------
extern "C" void kernel_launch(void* const* d_in, const int* in_sizes, int n_in,
                              void* d_out, int out_size, void* d_ws, size_t ws_size,
                              hipStream_t stream)
{
    const float* x1    = (const float*)d_in[0];
    const float* x2    = (const float*)d_in[1];
    const float* e1    = (const float*)d_in[2];
    const float* e2    = (const float*)d_in[3];
    const float* Wp    = (const float*)d_in[4];
    const float* bp    = (const float*)d_in[5];
    const float* gamma = (const float*)d_in[6];
    const float* beta  = (const float*)d_in[7];
    const float* W0 = (const float*)d_in[8],  *b0 = (const float*)d_in[9];
    const float* S0 = (const float*)d_in[10], *c0 = (const float*)d_in[11];
    const float* W1 = (const float*)d_in[12], *b1 = (const float*)d_in[13];
    const float* S1 = (const float*)d_in[14], *c1 = (const float*)d_in[15];
    const float* W2 = (const float*)d_in[16], *b2 = (const float*)d_in[17];
    const float* S2 = (const float*)d_in[18], *c2 = (const float*)d_in[19];
    const float* Wc = (const float*)d_in[20], *bc = (const float*)d_in[21];
    const float* binv = (const float*)d_in[22];
    const int* src1 = (const int*)d_in[23];
    const int* dst1 = (const int*)d_in[24];
    const int* src2 = (const int*)d_in[25];
    const int* dst2 = (const int*)d_in[26];

    float* ws    = (float*)d_ws;
    float* H     = ws;                         // 4096*256
    float* stats = H + 4096 * 256;             // 2048
    float* Kp    = stats + 2048;               // 8*4096
    float* Ke    = Kp + BB * NNQ;              // 8*36864
    float* xA    = Ke + BB * EE * EE;          // 8*4096*17
    float* xB    = xA + (size_t)BB * NNQ * 17; // 8*4096*17
    float* v3    = xB + (size_t)BB * NNQ * 17; // 8*4096
    float* ls    = v3 + BB * NNQ;              // 8*4225
    float* ssf   = ls + BB * 4225;             // 8*4096

    // 1) projection GEMM (WMMA bf16) for all four inputs at once
    proj_gemm_kernel<<<dim3(32, 8), 256, 0, stream>>>(x1, x2, e1, e2, Wp, bp, H);
    // 2) batch-norm stats + 3) normalize/ReLU/L2
    stats_kernel<<<4, 256, 0, stream>>>(H, stats);
    norm_kernel<<<4096, 256, 0, stream>>>(H, stats, gamma, beta);
    // 4) similarity GEMMs (WMMA): Kp = p1 p2^T, Ke = 0.5 q1 q2^T
    xyT_kernel<<<dim3(BB, 4, 4),   32, 0, stream>>>(H, 0,    512,  64,  1.0f, Kp);
    xyT_kernel<<<dim3(BB, 12, 12), 32, 0, stream>>>(H, 1024, 2560, 192, 0.5f, Ke);
    // 5) initial x = Kp^T.ravel()
    x0_kernel<<<BB, 256, 0, stream>>>(Kp, xA);
    // 6) three rounds of sparse message passing + Sinkhorn (ping-pong x)
    msg_mlp_kernel<<<dim3(NNQ, BB), 256, 0, stream>>>(Ke, Kp, src1, dst1, src2, dst2,
                                                      xA, xB, v3, W0, b0, S0, c0, 1);
    sinkhorn64_kernel<<<BB, 64, 0, stream>>>(v3, xB);
    msg_mlp_kernel<<<dim3(NNQ, BB), 256, 0, stream>>>(Ke, Kp, src1, dst1, src2, dst2,
                                                      xB, xA, v3, W1, b1, S1, c1, 17);
    sinkhorn64_kernel<<<BB, 64, 0, stream>>>(v3, xA);
    msg_mlp_kernel<<<dim3(NNQ, BB), 256, 0, stream>>>(Ke, Kp, src1, dst1, src2, dst2,
                                                      xA, xB, v3, W2, b2, S2, c2, 17);
    sinkhorn64_kernel<<<BB, 64, 0, stream>>>(v3, xB);
    // 7) readout + padded 65x65 Sinkhorn + global-max rescale
    final_v_kernel<<<BB, 256, 0, stream>>>(xB, Wc, bc, binv, ls);
    sinkhorn65_kernel<<<BB, 96, 0, stream>>>(ls, ssf);
    finalize_kernel<<<1, 256, 0, stream>>>(ssf, (float*)d_out);
}
